// NeuroEmoDynamics_64914135712035
// MI455X (gfx1250) — compile-verified, matrix operand order checked by reading, not run on hardware
//
#include <hip/hip_runtime.h>
#include <hip/hip_bf16.h>
#include <math.h>

typedef __attribute__((ext_vector_type(16))) _Float16 v16h;
typedef __attribute__((ext_vector_type(8)))  float    v8f;

#define B 2048

__device__ __forceinline__ float sigf(float x) { return 1.f / (1.f + expf(-x)); }

// ---------------------------------------------------------------------------
// Weight pre-pack: W[K x N] f32 row-major  ->  Wp in WMMA B-fragment order, f16.
// Fragment element e (0..15) of lane l for k-tile kt / n-tile nt holds
//   k = kt*32 + ((e<8)?0:16) + (l>>4)*8 + (e&7),  n = nt*16 + (l&15)
// Layout: Wp[(((nt*KT + kt)*32 + l)*16 + e)]  => each lane's 32B fragment is
// contiguous, so the GEMM loads it with two global_load_b128.
// Rows k >= Kreal are zero-padded (handles Wt1's K=100 -> 128).
// ---------------------------------------------------------------------------
__global__ void pack_w_kernel(const float* __restrict__ W, _Float16* __restrict__ Wp,
                              int Kreal, int Kp, int N)
{
    int tid = blockIdx.x * blockDim.x + threadIdx.x;
    if (tid >= Kp * N) return;
    int e = tid & 15;
    int l = (tid >> 4) & 31;
    int t = tid >> 9;                 // nt*KT + kt
    int KT = Kp >> 5;
    int kt = t % KT, nt = t / KT;
    int n = nt * 16 + (l & 15);
    int k = kt * 32 + ((e < 8) ? 0 : 16) + (l >> 4) * 8 + (e & 7);
    float v = (k < Kreal) ? W[k * N + n] : 0.f;
    Wp[tid] = (_Float16)v;
}

// ---------------------------------------------------------------------------
// WMMA GEMM: C[M x N] = act(A[M x K] @ W[K x N] + bias)
// One wave per block; each wave computes a 16x64 strip of C (4 accumulators).
// Requires K % 32 == 0, N % 64 == 0, M % 16 == 0. Branch-free inner loop.
// act: 0=none 1=sigmoid 2=relu 3=1+sigmoid
// ---------------------------------------------------------------------------
__global__ void wmma_gemm_kernel(const float* __restrict__ A, int lda,
                                 const _Float16* __restrict__ Wp,
                                 const float* __restrict__ bias,
                                 float* __restrict__ C, int ldc,
                                 int K, int act)
{
    const int lane = threadIdx.x;           // 0..31
    const int h    = lane >> 4;             // K-half group
    const int l16  = lane & 15;
    const int tileM  = blockIdx.y;
    const int ntBase = blockIdx.x * 4;
    const int KT = K >> 5;

    const float* Arow = A + (size_t)(tileM * 16 + l16) * lda;
    const size_t ntStride = (size_t)KT * 512;   // halfs between n-tiles

    v8f acc0 = {}, acc1 = {}, acc2 = {}, acc3 = {};

    for (int kt = 0; kt < KT; ++kt) {
        const int k0 = kt * 32;
        // A fragment: two contiguous 8-float runs per lane
        const float4* plo = (const float4*)(Arow + k0 + h * 8);
        const float4* phi = (const float4*)(Arow + k0 + 16 + h * 8);
        float4 a0 = plo[0], a1 = plo[1], a2 = phi[0], a3 = phi[1];
        float av[16] = { a0.x, a0.y, a0.z, a0.w, a1.x, a1.y, a1.z, a1.w,
                         a2.x, a2.y, a2.z, a2.w, a3.x, a3.y, a3.z, a3.w };
        v16h a;
#pragma unroll
        for (int e = 0; e < 16; ++e) a[e] = (_Float16)av[e];

        const _Float16* wbase = Wp + (((size_t)ntBase * KT + kt) * 32 + lane) * 16;
        v16h b0 = *(const v16h*)(wbase);
        v16h b1 = *(const v16h*)(wbase + ntStride);
        v16h b2 = *(const v16h*)(wbase + 2 * ntStride);
        v16h b3 = *(const v16h*)(wbase + 3 * ntStride);

        acc0 = __builtin_amdgcn_wmma_f32_16x16x32_f16(false, a, false, b0, (short)0, acc0, false, false);
        acc1 = __builtin_amdgcn_wmma_f32_16x16x32_f16(false, a, false, b1, (short)0, acc1, false, false);
        acc2 = __builtin_amdgcn_wmma_f32_16x16x32_f16(false, a, false, b2, (short)0, acc2, false, false);
        acc3 = __builtin_amdgcn_wmma_f32_16x16x32_f16(false, a, false, b3, (short)0, acc3, false, false);
    }

    // Epilogue: C fragment layout — VGPR i holds row tileM*16 + h*8 + i, col = n-tile base + l16
#pragma unroll
    for (int nt = 0; nt < 4; ++nt) {
        v8f acc = (nt == 0) ? acc0 : (nt == 1) ? acc1 : (nt == 2) ? acc2 : acc3;
        const int c = (ntBase + nt) * 16 + l16;
        const float bv = bias[c];
#pragma unroll
        for (int i = 0; i < 8; ++i) {
            int r = tileM * 16 + h * 8 + i;
            float v = acc[i] + bv;
            if (act == 1)      v = sigf(v);
            else if (act == 2) v = fmaxf(v, 0.f);
            else if (act == 3) v = 1.f + sigf(v);
            C[(size_t)r * ldc + c] = v;
        }
    }
}

// --------------------------- elementwise / gather ---------------------------

__global__ void gather_profile_kernel(const float* __restrict__ emb,
                                      const int* __restrict__ ids,
                                      float* __restrict__ prof)
{
    int tid = blockIdx.x * blockDim.x + threadIdx.x;
    if (tid >= B * 256) return;
    int b = tid >> 8, c = tid & 255;
    prof[tid] = emb[ids[b] * 256 + c];
}

// mean-pool token embeddings into a zero-padded [B,128] buffer (cols 100..127 = 0)
__global__ void tok_mean_kernel(const float* __restrict__ tok_emb,
                                const int* __restrict__ text,
                                float* __restrict__ emb)
{
    int tid = blockIdx.x * blockDim.x + threadIdx.x;
    if (tid >= B * 128) return;
    int b = tid >> 7, d = tid & 127;
    float r = 0.f;
    if (d < 100) {
        float s = 0.f;
        for (int t = 0; t < 128; ++t)
            s += tok_emb[text[b * 128 + t] * 100 + d];
        r = s * (1.f / 128.f);
    }
    emb[tid] = r;
}

__global__ void flag_kernel(const int* __restrict__ text, float* __restrict__ flag)
{
    int b = blockIdx.x * blockDim.x + threadIdx.x;
    if (b >= B) return;
    float f = 0.f;
    for (int t = 0; t < 128; ++t)
        if (text[b * 128 + t] < 8) f = 1.f;
    flag[b] = f;
}

// head6: out[b,j] = X[b,:1024] @ W[:,j] + bias[j]    (N=6, too thin for WMMA)
__global__ void head6_kernel(const float* __restrict__ X, const float* __restrict__ W,
                             const float* __restrict__ bias, float* __restrict__ out)
{
    int tid = blockIdx.x * blockDim.x + threadIdx.x;
    if (tid >= B * 6) return;
    int b = tid / 6, j = tid - b * 6;
    float s = bias[j];
    for (int k = 0; k < 1024; ++k)
        s += X[b * 1024 + k] * W[k * 6 + j];
    out[tid] = s;
}

__global__ void softmax6_kernel(const float* __restrict__ logits, float* __restrict__ pos)
{
    int b = blockIdx.x * blockDim.x + threadIdx.x;
    if (b >= B) return;
    float v[6], m = -1e30f;
    for (int j = 0; j < 6; ++j) { v[j] = logits[b * 6 + j]; m = fmaxf(m, v[j]); }
    float s = 0.f;
    for (int j = 0; j < 6; ++j) { v[j] = expf(v[j] - m); s += v[j]; }
    pos[b] = (v[1] + v[5]) / s;
}

// per-row: sample_scale = sig([text_enc,flag] @ W_scale + b) ; self_ref = sig(text_enc @ W_sr + b)
__global__ void scale_sr_kernel(const float* __restrict__ te, const float* __restrict__ flag,
                                const float* __restrict__ W_scale, const float* __restrict__ b_scale,
                                const float* __restrict__ W_sr, const float* __restrict__ b_sr,
                                float* __restrict__ sscale, float* __restrict__ out_sr)
{
    __shared__ float s1[256], s2[256];
    int b = blockIdx.x, t = threadIdx.x;
    float a1 = 0.f, a2 = 0.f;
    for (int k = t; k < 1024; k += 256) {
        float x = te[b * 1024 + k];
        a1 += x * W_scale[k];
        a2 += x * W_sr[k];
    }
    if (t == 0) a1 += flag[b] * W_scale[1024];
    s1[t] = a1; s2[t] = a2; __syncthreads();
    for (int off = 128; off > 0; off >>= 1) {
        if (t < off) { s1[t] += s1[t + off]; s2[t] += s2[t + off]; }
        __syncthreads();
    }
    if (t == 0) {
        sscale[b] = sigf(s1[0] + b_scale[0]);
        out_sr[b] = sigf(s2[0] + b_sr[0]);
    }
}

__global__ void textmod_kernel(const float* __restrict__ te, const float* __restrict__ gate,
                               const float* __restrict__ da, const float* __restrict__ sscale,
                               const float* __restrict__ sr, float* __restrict__ tm)
{
    int tid = blockIdx.x * blockDim.x + threadIdx.x;
    if (tid >= B * 1024) return;
    int b = tid >> 10;
    tm[tid] = te[tid] * (gate[tid] + da[tid] * (sscale[b] * sr[b]));
}

// 20-step sensory LIF (V_th=1.2, tau=30); only final spike consumed.
__global__ void pfc_lif_kernel(const float* __restrict__ sensory, const float* __restrict__ ser,
                               const float* __restrict__ pfb, float* __restrict__ pfc_out)
{
    int tid = blockIdx.x * blockDim.x + threadIdx.x;
    if (tid >= B * 512) return;
    const float decay = 0.96721648f;  // exp(-1/30)
    float V = 0.f, s = 0.f;
    for (int t = 0; t < 20; ++t) {
        float Vp = V * decay + sensory[t * (B * 512) + tid];
        s = (Vp > 1.2f) ? 1.f : 0.f;
        V = Vp * (1.f - s);
    }
    pfc_out[tid] = s * ser[tid] + pfb[tid];
}

__global__ void amyg_scale_kernel(float* __restrict__ integ_in, const float* __restrict__ ne)
{
    int tid = blockIdx.x * blockDim.x + threadIdx.x;
    if (tid >= B * 256) return;
    int b = tid >> 8, c = tid & 255;
    integ_in[b * 1792 + c] *= ne[tid];
}

// buf_ta = [thal, amyg] ; also copies prev_feedback[:, :256] into integ_in cols 1536..1791
__global__ void build_concat_kernel(const float* __restrict__ integ_in_ro,
                                    float* __restrict__ buf_ta,
                                    float* __restrict__ integ_in,
                                    const float* __restrict__ pfb)
{
    int tid = blockIdx.x * blockDim.x + threadIdx.x;
    if (tid >= B * 512) return;
    int b = tid >> 9, c = tid & 511;
    buf_ta[tid] = (c < 256) ? integ_in_ro[b * 1792 + 512 + c]
                            : integ_in_ro[b * 1792 + (c - 256)];
    if (c < 256)
        integ_in[b * 1792 + 1536 + c] = pfb[b * 512 + c];
}

__global__ void layernorm_kernel(float* __restrict__ x, const float* __restrict__ g,
                                 const float* __restrict__ bb)
{
    __shared__ float s1[256], s2[256];
    int b = blockIdx.x, t = threadIdx.x;
    float a = 0.f, a2 = 0.f;
    for (int k = t; k < 1024; k += 256) {
        float v = x[b * 1024 + k]; a += v; a2 += v * v;
    }
    s1[t] = a; s2[t] = a2; __syncthreads();
    for (int off = 128; off > 0; off >>= 1) {
        if (t < off) { s1[t] += s1[t + off]; s2[t] += s2[t + off]; }
        __syncthreads();
    }
    float mu = s1[0] * (1.f / 1024.f);
    float var = s2[0] * (1.f / 1024.f) - mu * mu;
    float inv = rsqrtf(var + 1e-5f);
    for (int k = t; k < 1024; k += 256)
        x[b * 1024 + k] = (x[b * 1024 + k] - mu) * inv * g[k] + bb[k];
}

// override / fuse-gate / fused output, one block per row
__global__ void ovfg_kernel(const float* __restrict__ te, const float* __restrict__ prof,
                            const float* __restrict__ sr, const float* __restrict__ pos,
                            const float* __restrict__ W_ov, const float* __restrict__ b_ov,
                            const float* __restrict__ tm, const float* __restrict__ io,
                            const float* __restrict__ W_fg, const float* __restrict__ b_fg,
                            float* __restrict__ fused)
{
    __shared__ float s1[256], s2[256];
    __shared__ float wsh;
    int b = blockIdx.x, t = threadIdx.x;
    float a1 = 0.f, a2 = 0.f;
    for (int k = t; k < 1024; k += 256) {
        a1 += te[b * 1024 + k] * W_ov[k];
        a2 += tm[b * 1024 + k] * W_fg[k] + io[b * 1024 + k] * W_fg[1024 + k];
    }
    a1 += prof[b * 256 + t] * W_ov[1024 + t];
    if (t == 0)  a1 += sr[b] * W_ov[1280] + pos[b] * W_ov[1281];
    s1[t] = a1; s2[t] = a2; __syncthreads();
    for (int off = 128; off > 0; off >>= 1) {
        if (t < off) { s1[t] += s1[t + off]; s2[t] += s2[t + off]; }
        __syncthreads();
    }
    if (t == 0) {
        float ov = sigf(s1[0] + b_ov[0]);
        float w  = sigf(s2[0] + b_fg[0]);
        wsh = fmaxf(w, ov);
    }
    __syncthreads();
    float w = wsh;
    for (int k = t; k < 1024; k += 256)
        fused[b * 1024 + k] = w * tm[b * 1024 + k] + (1.f - w) * io[b * 1024 + k];
}

// 10-step striatum LIF (V_th=1.0, tau=20); writes spikes + voltages per step.
__global__ void striatum_kernel(const float* __restrict__ io, const float* __restrict__ reward,
                                const float* __restrict__ da,
                                float* __restrict__ spks, float* __restrict__ volts)
{
    int tid = blockIdx.x * blockDim.x + threadIdx.x;
    if (tid >= B * 1024) return;
    int b = tid >> 10;
    const float decay = 0.95122945f;  // exp(-1/20)
    float I = io[tid] * reward[b] * da[tid];
    float V = 0.f;
    for (int t = 0; t < 10; ++t) {
        float Vp = V * decay + I;
        float s = (Vp > 1.f) ? 1.f : 0.f;
        spks[t * (B * 1024) + tid] = s;
        volts[t * (B * 1024) + tid] = Vp;
        V = Vp * (1.f - s);
    }
}

// ---------------------------------------------------------------------------

extern "C" void kernel_launch(void* const* d_in, const int* in_sizes, int n_in,
                              void* d_out, int out_size, void* d_ws, size_t ws_size,
                              hipStream_t stream)
{
    (void)in_sizes; (void)n_in; (void)out_size; (void)ws_size;

    const float* profile_emb = (const float*)d_in[0];
    const float* W_ser  = (const float*)d_in[1];
    const float* W_ne   = (const float*)d_in[2];
    const float* W_da   = (const float*)d_in[3];
    const float* tok_emb= (const float*)d_in[4];
    const float* Wt1    = (const float*)d_in[5];
    const float* Wt2    = (const float*)d_in[6];
    const float* W_gate = (const float*)d_in[7];
    const float* W_scale= (const float*)d_in[8];
    const float* W_sr   = (const float*)d_in[9];
    const float* W_ov   = (const float*)d_in[10];
    const float* W_pa   = (const float*)d_in[11];
    const float* W_ph   = (const float*)d_in[12];
    const float* W_pt   = (const float*)d_in[13];
    const float* W_ah   = (const float*)d_in[14];
    const float* W_ht   = (const float*)d_in[15];
    const float* W_ta   = (const float*)d_in[16];
    const float* W_int  = (const float*)d_in[17];
    /* W_fb d_in[18] dead: _new_fb never returned */
    const float* W_fg   = (const float*)d_in[19];
    const float* W_cls  = (const float*)d_in[20];
    const float* b_ser  = (const float*)d_in[21];
    const float* b_ne   = (const float*)d_in[22];
    const float* b_da   = (const float*)d_in[23];
    const float* bt1    = (const float*)d_in[24];
    const float* bt2    = (const float*)d_in[25];
    const float* b_gate = (const float*)d_in[26];
    const float* b_scale= (const float*)d_in[27];
    const float* b_sr   = (const float*)d_in[28];
    const float* b_ov   = (const float*)d_in[29];
    const float* b_pa   = (const float*)d_in[30];
    const float* b_ph   = (const float*)d_in[31];
    const float* b_pt   = (const float*)d_in[32];
    const float* b_ah   = (const float*)d_in[33];
    const float* b_ht   = (const float*)d_in[34];
    const float* b_ta   = (const float*)d_in[35];
    const float* b_int  = (const float*)d_in[36];
    /* b_fb d_in[37] dead */
    const float* b_fg   = (const float*)d_in[38];
    const float* b_cls  = (const float*)d_in[39];
    const float* ln_t_g = (const float*)d_in[40];
    const float* ln_t_b = (const float*)d_in[41];
    const float* ln_i_g = (const float*)d_in[42];
    const float* ln_i_b = (const float*)d_in[43];
    const float* sensory = (const float*)d_in[44];
    const float* reward  = (const float*)d_in[45];
    const int*   text    = (const int*)d_in[46];
    const int*   prof_ids= (const int*)d_in[47];
    const float* prev_fb = (const float*)d_in[48];

    // output regions (return order: spks, volts, logits, aux, ser, da, ne, self_ref)
    float* out = (float*)d_out;
    float* out_spks   = out;
    float* out_volts  = out_spks  + 10 * B * 1024;
    float* out_logits = out_volts + 10 * B * 1024;
    float* out_aux    = out_logits + B * 6;
    float* out_ser    = out_aux    + B * 6;
    float* out_da     = out_ser    + B * 512;
    float* out_ne     = out_da     + B * 1024;
    float* out_sr     = out_ne     + B * 256;

    // workspace layout (f32 activations)
    float* ws = (float*)d_ws;
    float* prof     = ws;                       // B*256
    float* emb      = prof     + B * 256;       // B*128 (zero-padded from 100)
    float* flagbuf  = emb      + B * 128;       // B
    float* h1       = flagbuf  + B;             // B*128
    float* text_enc = h1       + B * 128;       // B*1024
    float* gate     = text_enc + B * 1024;      // B*1024
    float* sscale   = gate     + B * 1024;      // B
    float* posscore = sscale   + B;             // B
    float* text_mod = posscore + B;             // B*1024
    float* pfc_out  = text_mod + B * 1024;      // B*512
    float* integ_in = pfc_out  + B * 512;       // B*1792
    float* buf_ta   = integ_in + B * 1792;      // B*512
    float* integ_out= buf_ta   + B * 512;       // B*1024 (LN in place)
    float* fused    = integ_out+ B * 1024;      // B*1024

    // packed f16 weights (WMMA B-fragment order)
    _Float16* hp = (_Float16*)(fused + B * 1024);
    _Float16* p_ser  = hp;                        // 256*512
    _Float16* p_ne   = p_ser  + 256 * 512;        // 256*256
    _Float16* p_da   = p_ne   + 256 * 256;        // 256*1024
    _Float16* p_t1   = p_da   + 256 * 1024;       // 128*128 (padded)
    _Float16* p_t2   = p_t1   + 128 * 128;        // 128*1024
    _Float16* p_gate = p_t2   + 128 * 1024;       // 1024*1024
    _Float16* p_pa   = p_gate + 1024 * 1024;      // 512*256
    _Float16* p_ph   = p_pa   + 512 * 256;
    _Float16* p_pt   = p_ph   + 512 * 256;
    _Float16* p_ah   = p_pt   + 512 * 256;
    _Float16* p_ht   = p_ah   + 512 * 256;
    _Float16* p_ta   = p_ht   + 512 * 256;
    _Float16* p_int  = p_ta   + 512 * 256;        // 1792*1024

    const int T256 = 256;
    auto blocks = [](int n) { return (n + 255) / 256; };
    enum { ACT_NONE = 0, ACT_SIG = 1, ACT_RELU = 2, ACT_1PSIG = 3 };

    auto pack = [&](const float* W, _Float16* Wp, int Kreal, int Kp, int N) {
        pack_w_kernel<<<blocks(Kp * N), T256, 0, stream>>>(W, Wp, Kreal, Kp, N);
    };
    auto gemm = [&](const float* A, int lda, const _Float16* Wp, int N, const float* bias,
                    float* C, int ldc, int K, int act) {
        dim3 grid(N / 64, B / 16);
        wmma_gemm_kernel<<<grid, 32, 0, stream>>>(A, lda, Wp, bias, C, ldc, K, act);
    };

    // ---- pack all WMMA-path weights (f32 -> f16 fragment order) ----
    pack(W_ser, p_ser, 256, 256, 512);
    pack(W_ne,  p_ne,  256, 256, 256);
    pack(W_da,  p_da,  256, 256, 1024);
    pack(Wt1,   p_t1,  100, 128, 128);
    pack(Wt2,   p_t2,  128, 128, 1024);
    pack(W_gate,p_gate,1024,1024,1024);
    pack(W_pa,  p_pa,  512, 512, 256);
    pack(W_ph,  p_ph,  512, 512, 256);
    pack(W_pt,  p_pt,  512, 512, 256);
    pack(W_ah,  p_ah,  512, 512, 256);
    pack(W_ht,  p_ht,  512, 512, 256);
    pack(W_ta,  p_ta,  512, 512, 256);
    pack(W_int, p_int, 1792,1792,1024);

    // ---- profile branch ----
    gather_profile_kernel<<<blocks(B * 256), T256, 0, stream>>>(profile_emb, prof_ids, prof);
    gemm(prof, 256, p_ser, 512, b_ser, out_ser, 512, 256, ACT_SIG);
    gemm(prof, 256, p_ne, 256, b_ne, out_ne, 256, 256, ACT_1PSIG);
    gemm(prof, 256, p_da, 1024, b_da, out_da, 1024, 256, ACT_SIG);

    // ---- text encoder ----
    tok_mean_kernel<<<blocks(B * 128), T256, 0, stream>>>(tok_emb, text, emb);
    flag_kernel<<<blocks(B), T256, 0, stream>>>(text, flagbuf);
    gemm(emb, 128, p_t1, 128, bt1, h1, 128, 128, ACT_RELU);
    gemm(h1, 128, p_t2, 1024, bt2, text_enc, 1024, 128, ACT_NONE);

    // ---- aux head, gating scalars ----
    head6_kernel<<<blocks(B * 6), T256, 0, stream>>>(text_enc, W_cls, b_cls, out_aux);
    softmax6_kernel<<<blocks(B), T256, 0, stream>>>(out_aux, posscore);
    gemm(text_enc, 1024, p_gate, 1024, b_gate, gate, 1024, 1024, ACT_SIG);
    scale_sr_kernel<<<B, 256, 0, stream>>>(text_enc, flagbuf, W_scale, b_scale,
                                           W_sr, b_sr, sscale, out_sr);
    textmod_kernel<<<blocks(B * 1024), T256, 0, stream>>>(text_enc, gate, out_da,
                                                          sscale, out_sr, text_mod);

    // ---- sensory LIF -> limbic projections (assembled directly into [B,1792]) ----
    pfc_lif_kernel<<<blocks(B * 512), T256, 0, stream>>>(sensory, out_ser, prev_fb, pfc_out);
    gemm(pfc_out, 512, p_pa, 256, b_pa, integ_in + 0,    1792, 512, ACT_NONE); // amyg
    gemm(pfc_out, 512, p_ph, 256, b_ph, integ_in + 256,  1792, 512, ACT_NONE); // hipp
    gemm(pfc_out, 512, p_pt, 256, b_pt, integ_in + 512,  1792, 512, ACT_NONE); // thal
    amyg_scale_kernel<<<blocks(B * 256), T256, 0, stream>>>(integ_in, out_ne);
    build_concat_kernel<<<blocks(B * 512), T256, 0, stream>>>(integ_in, buf_ta,
                                                              integ_in, prev_fb);
    gemm(integ_in + 0,   1792, p_ah, 256, b_ah, integ_in + 768,  1792, 512, ACT_NONE);
    gemm(integ_in + 256, 1792, p_ht, 256, b_ht, integ_in + 1024, 1792, 512, ACT_NONE);
    gemm(buf_ta, 512,          p_ta, 256, b_ta, integ_in + 1280, 1792, 512, ACT_NONE);
    gemm(integ_in, 1792, p_int, 1024, b_int, integ_out, 1024, 1792, ACT_NONE);

    // ---- layernorms, fusion, heads ----
    layernorm_kernel<<<B, 256, 0, stream>>>(text_mod, ln_t_g, ln_t_b);
    layernorm_kernel<<<B, 256, 0, stream>>>(integ_out, ln_i_g, ln_i_b);
    ovfg_kernel<<<B, 256, 0, stream>>>(text_enc, prof, out_sr, posscore, W_ov, b_ov,
                                       text_mod, integ_out, W_fg, b_fg, fused);
    head6_kernel<<<blocks(B * 6), T256, 0, stream>>>(fused, W_cls, b_cls, out_logits);

    // ---- striatum LIF (dominant output traffic, pure streaming) ----
    striatum_kernel<<<blocks(B * 1024), T256, 0, stream>>>(integ_out, reward, out_da,
                                                           out_spks, out_volts);
}
